// Layer_8022998909164
// MI455X (gfx1250) — compile-verified, matrix operand order checked by reading, not run on hardware
//
#include <hip/hip_runtime.h>
#include <hip/hip_bf16.h>

// ---------------------------------------------------------------------------
// Shapes: x (T=8, B=32, Cin=64, H=32, W=32); conv_w (128, 64, 3, 3)
// N = T*B = 256 images, Cout = 128, HW = 1024, K = Cin*9 = 576.
// Conv as implicit GEMM with v_wmma_f32_16x16x32_bf16 (fp32 accumulate).
// Each block: 128 Cout x 128 pixels (= 4 rows of one image). The block's
// receptive field (6 rows x 34 cols x 64 ci, halo pre-zeroed) is staged in
// LDS as bf16 once; B fragments are read straight out of it per tap.
// conv_b cancels exactly under training-mode BatchNorm -> unused.
// ---------------------------------------------------------------------------

typedef __attribute__((ext_vector_type(16))) __bf16    v16bf;
typedef __attribute__((ext_vector_type(8)))  float     v8f;
typedef __attribute__((ext_vector_type(4)))  unsigned  v4u;

union FragBF {
    v16bf v;
    v4u   u[2];
};

#define N_IMG   256
#define CIN     64
#define COUT    128
#define HWSZ    1024
#define NHW     (N_IMG * HWSZ)           // 262144 pixels total
#define TPLANE  (32 * COUT * HWSZ)       // 4,194,304 elements per timestep
#define XSTRIDE 36                       // dwords per 64-ci bf16 row (padded 144B)
#define XROWS   (6 * 34)                 // 204 spatial rows in the x-tile

__device__ __forceinline__ unsigned pack_bf16(float a, float b) {
    unsigned ua = __float_as_uint(a);
    unsigned ub = __float_as_uint(b);
    ua += 0x7FFFu + ((ua >> 16) & 1u);   // round-to-nearest-even
    ub += 0x7FFFu + ((ub >> 16) & 1u);
    return (ua >> 16) | (ub & 0xFFFF0000u);
}

// ---------------------------------------------------------------------------
// Kernel 0: weights OIHW fp32 -> [tap][co][ci] bf16
// ---------------------------------------------------------------------------
__global__ __launch_bounds__(256) void wtrans_kernel(
    const float* __restrict__ w, unsigned short* __restrict__ wT)
{
    int idx = blockIdx.x * 256 + threadIdx.x;      // total 9*128*64 = 73728
    if (idx >= 9 * COUT * CIN) return;
    int tap = idx >> 13;                           // / 8192
    int rem = idx & 8191;
    int co  = rem >> 6;
    int ci  = rem & 63;
    unsigned u = __float_as_uint(w[co * 576 + ci * 9 + tap]);
    u += 0x7FFFu + ((u >> 16) & 1u);
    wT[idx] = (unsigned short)(u >> 16);
}

// ---------------------------------------------------------------------------
// Kernel 1: implicit-GEMM conv.
// ---------------------------------------------------------------------------
__global__ __launch_bounds__(256) void conv_wmma_kernel(
    const float* __restrict__ x,          // [256][64][1024] fp32
    const unsigned* __restrict__ wT,      // bf16 pairs: [9][128][64] as dwords
    float* __restrict__ y)                // [256][128][1024] fp32
{
    __shared__ unsigned ldsX[XROWS * XSTRIDE];   // 29,376 B  (x receptive field)
    __shared__ unsigned ldsA[COUT  * XSTRIDE];   // 18,432 B  (one tap's weights)

    const int tid  = threadIdx.x;
    const int wave = tid >> 5;
    const int lane = tid & 31;
    const int hf   = lane >> 4;        // K-half select per ISA 16-bit A/B layout
    const int ln   = lane & 15;        // row (A: M=co) / col (B: N=px)

    const int px0 = blockIdx.x << 7;   // 128 pixels per block
    const int n   = px0 >> 10;         // image index (block fits in one image)
    const int h0  = (px0 & 1023) >> 5; // first of the block's 4 rows

    const int wco = (wave & 1) * 64;   // wave's Cout base
    const int wpx = (wave >> 1) * 32;  // wave's pixel base

    // ---- stage x receptive field: zero halo, then fill interior ----
    for (int i = tid; i < XROWS * XSTRIDE; i += 256) ldsX[i] = 0u;
    __syncthreads();
    for (int m = tid; m < 6 * 32 * 32; m += 256) {   // 6144 dwords (2 ci each)
        const int w   = m & 31;                      // coalesced in w
        const int ci2 = (m >> 5) & 31;               // ci pair index
        const int hr  = m >> 10;                     // 0..5 tile row
        const int h   = h0 - 1 + hr;
        if (h >= 0 && h < 32) {
            const float* src = x + ((n * CIN + ci2 * 2) << 10) + (h << 5) + w;
            float v0 = src[0];
            float v1 = src[HWSZ];
            ldsX[(hr * 34 + w + 1) * XSTRIDE + ci2] = pack_bf16(v0, v1);
        }
    }

    v8f acc[4][2] = {};

    // per-lane B row base inside the padded tile (h+1, w+1 offsets for halo)
    // used as: rowB = baseB[j] + dh*34 + dw
    int baseB[2];
    #pragma unroll
    for (int j = 0; j < 2; ++j) {
        const int pl = wpx + j * 16 + ln;            // 0..127
        baseB[j] = ((pl >> 5) + 1) * 34 + (pl & 31) + 1;
    }

    for (int tap = 0; tap < 9; ++tap) {
        const int dh = tap / 3 - 1;
        const int dw = tap % 3 - 1;

        __syncthreads();   // previous tap's A readers are done
        // ---- stage A: 128 co x 64 ci bf16 for this tap ----
        {
            const int co  = tid >> 1;
            const int cof = (tid & 1) * 16;          // dword offset in row
            const unsigned* src = wT + tap * (COUT * CIN / 2) + co * 32 + cof;
            #pragma unroll
            for (int jj = 0; jj < 4; ++jj) {
                v4u t = *(const v4u*)(src + jj * 4);
                *(v4u*)&ldsA[co * XSTRIDE + cof + jj * 4] = t;
            }
        }
        if (tap < 8)   // gfx1250 global_prefetch of next tap's weights
            __builtin_prefetch(wT + (tap + 1) * (COUT * CIN / 2), 0, 1);
        __syncthreads();

        #pragma unroll
        for (int cb = 0; cb < 2; ++cb) {
            // fragment chunk offsets (dwords): 16-bit layout -> half hf holds
            // K hf*8..hf*8+7 and 16+hf*8..23+hf*8 of the 32-wide K block
            const int c0 = cb * 16 + hf * 4;
            const int c1 = c0 + 8;

            FragBF bfrag[2];
            #pragma unroll
            for (int j = 0; j < 2; ++j) {
                const unsigned* row =
                    &ldsX[(baseB[j] + dh * 34 + dw) * XSTRIDE];
                bfrag[j].u[0] = *(const v4u*)(row + c0);
                bfrag[j].u[1] = *(const v4u*)(row + c1);
            }
            #pragma unroll
            for (int i = 0; i < 4; ++i) {
                FragBF a;
                const unsigned* row = &ldsA[(wco + i * 16 + ln) * XSTRIDE];
                a.u[0] = *(const v4u*)(row + c0);
                a.u[1] = *(const v4u*)(row + c1);
                #pragma unroll
                for (int j = 0; j < 2; ++j) {
                    acc[i][j] = __builtin_amdgcn_wmma_f32_16x16x32_bf16(
                        false, a.v, false, bfrag[j].v,
                        (short)0, acc[i][j], false, false);
                }
            }
        }
    }

    // ---- writeback: C/D layout: lane ln = col(px), VGPR r -> row co = r+8*hf
    #pragma unroll
    for (int i = 0; i < 4; ++i) {
        const int co = wco + i * 16 + hf * 8;
        #pragma unroll
        for (int j = 0; j < 2; ++j) {
            const int hw2 = (px0 & 1023) + wpx + j * 16 + ln;
            float* dst = y + ((n * COUT) << 10) + hw2;
            #pragma unroll
            for (int r = 0; r < 8; ++r)
                dst[(co + r) << 10] = acc[i][j][r];
        }
    }
}

// ---------------------------------------------------------------------------
// Kernel 2: per-channel mean / var over (N, H, W)
// ---------------------------------------------------------------------------
__global__ __launch_bounds__(256) void stats_kernel(
    const float* __restrict__ y, float* __restrict__ stats)
{
    __shared__ float sh_s[256];
    __shared__ float sh_q[256];
    const int c = blockIdx.x;
    float s = 0.0f, q = 0.0f;
    for (int i = threadIdx.x; i < NHW; i += 256) {
        const int nn = i >> 10;
        const int hw = i & 1023;
        float v = y[nn * (COUT * HWSZ) + (c << 10) + hw];
        s += v;
        q += v * v;
    }
    sh_s[threadIdx.x] = s;
    sh_q[threadIdx.x] = q;
    __syncthreads();
    for (int off = 128; off > 0; off >>= 1) {
        if ((int)threadIdx.x < off) {
            sh_s[threadIdx.x] += sh_s[threadIdx.x + off];
            sh_q[threadIdx.x] += sh_q[threadIdx.x + off];
        }
        __syncthreads();
    }
    if (threadIdx.x == 0) {
        const float inv_n = 1.0f / (float)NHW;
        float mean = sh_s[0] * inv_n;
        float var  = sh_q[0] * inv_n - mean * mean;
        stats[c]        = mean;
        stats[COUT + c] = var;
    }
}

// ---------------------------------------------------------------------------
// Kernel 3: BN affine + LIF recurrence over T, emit spikes
// idx = b*131072 + c*1024 + hw ; y/out element for step t = t*TPLANE + idx
// ---------------------------------------------------------------------------
__global__ __launch_bounds__(256) void lif_kernel(
    const float* __restrict__ y, const float* __restrict__ stats,
    const float* __restrict__ gamma, const float* __restrict__ beta,
    float* __restrict__ out)
{
    const int idx = blockIdx.x * 256 + threadIdx.x;  // < 4,194,304
    const int c = (idx >> 10) & (COUT - 1);
    const float mean = stats[c];
    const float var  = stats[COUT + c];
    const float inv  = gamma[c] * rsqrtf(var + 1e-5f);
    const float shf  = beta[c] - mean * inv;

    float u = 0.0f;
    #pragma unroll
    for (int t = 0; t < 8; ++t) {
        float yv = y[t * TPLANE + idx];
        u = u * 0.99f + (yv * inv + shf);     // TAU = 0.99
        float sp = (u > 1.0f) ? 1.0f : 0.0f;  // heaviside(u - THRESH > 0)
        u -= sp;                               // spike * THRESH, THRESH = 1
        out[t * TPLANE + idx] = sp;
    }
}

// ---------------------------------------------------------------------------
extern "C" void kernel_launch(void* const* d_in, const int* in_sizes, int n_in,
                              void* d_out, int out_size, void* d_ws, size_t ws_size,
                              hipStream_t stream) {
    (void)in_sizes; (void)n_in; (void)out_size; (void)ws_size;
    const float* x      = (const float*)d_in[0];
    const float* conv_w = (const float*)d_in[1];
    // d_in[2] = conv_b: cancels exactly under training-mode BatchNorm.
    const float* gamma  = (const float*)d_in[3];
    const float* beta   = (const float*)d_in[4];
    float* out = (float*)d_out;

    char* ws = (char*)d_ws;
    float*          y     = (float*)ws;                              // 134,217,728 B
    unsigned short* wT    = (unsigned short*)(ws + 134217728ull);    //     147,456 B
    float*          stats = (float*)(ws + 134217728ull + 147456ull); //       1,024 B

    wtrans_kernel   <<<288,        256, 0, stream>>>(conv_w, wT);
    conv_wmma_kernel<<<NHW / 128,  256, 0, stream>>>(x, (const unsigned*)wT, y);
    stats_kernel    <<<COUT,       256, 0, stream>>>(y, stats);
    lif_kernel      <<<TPLANE/256, 256, 0, stream>>>(y, stats, gamma, beta, out);
}
